// PhaseFunctionedLinear_80762565034652
// MI455X (gfx1250) — matrix operand order repaired and checked by reading, not compiled, not run on hardware
//
#include <hip/hip_runtime.h>

// ---------------- problem constants ----------------
#define BATCH   8192
#define IDIM    1024
#define ODIM    1024
#define NCP     6
#define KTOT    6144          // NCP * IDIM
#define KPAD    6176          // + 32 bias/zero K-slots  (== 193 * BK)
#define KT_MAIN 192           // KTOT / BK
#define NKT     193

// ---------------- tiling ----------------
#define BM     128
#define BN     128
#define BK     32
#define PITCH  40             // bf16 elements per LDS row (32 + 8 pad -> conflict-free)

typedef unsigned short u16;
typedef __attribute__((ext_vector_type(16))) __bf16 bf16x16;
typedef __attribute__((ext_vector_type(8)))  float  f32x8;

struct __attribute__((aligned(8)))  U16x4 { u16 a, b, c, d; };
union  __attribute__((aligned(16))) FragU { uint4 q[2]; bf16x16 v; };

// ---------------------------------------------------------------------------
// fp32 -> bf16 hi/lo split (hi = truncation; lo = residual truncated)
// ---------------------------------------------------------------------------
__device__ __forceinline__ void splitbf(float x, u16& hi, u16& lo) {
    unsigned u  = __float_as_uint(x);
    unsigned hb = u & 0xffff0000u;
    hi = (u16)(hb >> 16);
    float rem = x - __uint_as_float(hb);
    lo = (u16)(__float_as_uint(rem) >> 16);
}

// ---------------------------------------------------------------------------
// Prep 1: Catmull-Rom spline weights -> dense alpha[b][8] (2 zero pads)
// ---------------------------------------------------------------------------
__global__ void pfnn_alpha_kernel(const float* __restrict__ phase,
                                  float* __restrict__ alpha) {
    int b = blockIdx.x * blockDim.x + threadIdx.x;
    if (b >= BATCH) return;
    const float two_pi = 6.283185307179586f;
    float ph = fmodf(phase[b], two_pi);
    if (ph < 0.0f) ph += two_pi;
    float pos  = ph * ((float)NCP / two_pi);
    float base = floorf(pos);
    int   bi   = (int)base;
    float t  = pos - base;
    float t2 = t * t;
    float t3 = t2 * t;
    float w0 = -0.5f * t + t2 - 0.5f * t3;
    float w1 = 1.0f - 2.5f * t2 + 1.5f * t3;
    float w2 = 0.5f * t + 2.0f * t2 - 1.5f * t3;
    float w3 = -0.5f * t2 + 0.5f * t3;
    float a[8] = {0.f, 0.f, 0.f, 0.f, 0.f, 0.f, 0.f, 0.f};
    a[((bi - 1) % NCP + NCP) % NCP] += w0;
    a[((bi    ) % NCP + NCP) % NCP] += w1;
    a[((bi + 1) % NCP + NCP) % NCP] += w2;
    a[((bi + 2) % NCP + NCP) % NCP] += w3;
    float4* dst = (float4*)(alpha + (size_t)b * 8);
    dst[0] = make_float4(a[0], a[1], a[2], a[3]);
    dst[1] = make_float4(a[4], a[5], a[6], a[7]);
}

// ---------------------------------------------------------------------------
// Prep 2: split W' (weights + bias K-slots) once into bf16 hi/lo, layout [o][k]
//         B'[o][k] : k<6144 -> W[k/1024, o, k%1024]; 6144<=k<6150 -> bias[k-6144,o]; else 0
// ---------------------------------------------------------------------------
__global__ void pfnn_presplit_kernel(const float* __restrict__ W,
                                     const float* __restrict__ bias,
                                     u16* __restrict__ bhi,
                                     u16* __restrict__ blo) {
    const int o = blockIdx.y;
    const int k = blockIdx.x * blockDim.x + threadIdx.x;
    if (k >= KPAD) return;
    float v;
    if (k < KTOT) {
        v = W[(size_t)(k >> 10) * (ODIM * IDIM) + (size_t)o * IDIM + (k & 1023)];
    } else {
        const int c = k - KTOT;
        v = (c < NCP) ? bias[(size_t)c * ODIM + o] : 0.f;
    }
    u16 h, l;
    splitbf(v, h, l);
    bhi[(size_t)o * KPAD + k] = h;
    blo[(size_t)o * KPAD + k] = l;
}

// ---------------------------------------------------------------------------
// A stage: global -> registers (fp32), alpha folded in.
// A'[b,k]: k<6144 -> alpha[b,k/1024]*x[b,k%1024]; k>=6144 -> alpha[b,k-6144]
// ---------------------------------------------------------------------------
__device__ __forceinline__ void load_a_stage(const float* __restrict__ x,
                                             const float* __restrict__ alpha,
                                             int bBase, int kt, int t, float* aSt) {
    const int rb = t >> 3;
    const int kk = (t & 7) << 2;
    if (kt < KT_MAIN) {
        const int c  = kt >> 5;
        const int i0 = (kt & 31) << 5;
#pragma unroll
        for (int p = 0; p < 4; ++p) {
            const int b = bBase + rb + p * 32;
            const float al = alpha[(size_t)b * 8 + c];
            const float4 v = *(const float4*)(x + (size_t)b * IDIM + i0 + kk);
            aSt[p * 4 + 0] = v.x * al;
            aSt[p * 4 + 1] = v.y * al;
            aSt[p * 4 + 2] = v.z * al;
            aSt[p * 4 + 3] = v.w * al;
        }
    } else {  // bias K-slots: A' carries alpha itself (rows padded with zeros)
#pragma unroll
        for (int p = 0; p < 4; ++p) {
            const int b = bBase + rb + p * 32;
            float4 v = make_float4(0.f, 0.f, 0.f, 0.f);
            if (kk < 8) v = *(const float4*)(alpha + (size_t)b * 8 + kk);
            aSt[p * 4 + 0] = v.x;
            aSt[p * 4 + 1] = v.y;
            aSt[p * 4 + 2] = v.z;
            aSt[p * 4 + 3] = v.w;
        }
    }
}

// ---------------------------------------------------------------------------
// B stage: CDNA5 async copy global(bf16, final layout) -> LDS, ASYNCcnt-tracked
// Each lane moves 32 contiguous bytes (2 x b128 via offset:16).
// ---------------------------------------------------------------------------
__device__ __forceinline__ void issue_async_b128x2(const u16* __restrict__ gbase,
                                                   const u16* lds_ptr,
                                                   unsigned gOff) {
    unsigned lAddr = (unsigned)(size_t)lds_ptr;   // low 32 bits = LDS byte offset
    asm volatile("global_load_async_to_lds_b128 %0, %1, %2\n\t"
                 "global_load_async_to_lds_b128 %0, %1, %2 offset:16"
                 :: "v"(lAddr), "v"(gOff), "s"(gbase)
                 : "memory");
}

__device__ __forceinline__ void stage_b_async(const u16* __restrict__ bhiG,
                                              const u16* __restrict__ bloG,
                                              u16* sBhiBuf, u16* sBloBuf,
                                              int nBase, int kt, int t) {
    const int n  = t >> 1;
    const int cp = t & 1;
    const unsigned gOff =
        (unsigned)(((size_t)(nBase + n) * KPAD + (size_t)kt * BK) * 2) + cp * 32;
    issue_async_b128x2(bhiG, sBhiBuf + n * PITCH + cp * 16, gOff);
    issue_async_b128x2(bloG, sBloBuf + n * PITCH + cp * 16, gOff);
}

// ---------------------------------------------------------------------------
// Fused GEMM: out[8192,1024] = A'(8192 x 6176) x B'(6176 x 1024), bf16x3 WMMA
// ---------------------------------------------------------------------------
__global__ __launch_bounds__(256)
void pfnn_gemm_kernel(const float* __restrict__ x,
                      const float* __restrict__ alpha,
                      const u16* __restrict__ bhiG,
                      const u16* __restrict__ bloG,
                      float* __restrict__ out) {
    __shared__ __attribute__((aligned(16))) u16 sAhi[BM * PITCH];
    __shared__ __attribute__((aligned(16))) u16 sAlo[BM * PITCH];
    __shared__ __attribute__((aligned(16))) u16 sBhi[2][BN * PITCH];
    __shared__ __attribute__((aligned(16))) u16 sBlo[2][BN * PITCH];

    const int t     = threadIdx.x;
    const int lane  = t & 31;
    const int wid   = t >> 5;
    const int waveMOff = (wid & 1) << 6;   // 0 or 64
    const int waveNOff = (wid >> 1) << 5;  // 0,32,64,96
    const int lh = lane >> 4;              // lane half (K/M-half selector)
    const int ll = lane & 15;

    const int bBase = blockIdx.y * BM;
    const int nBase = blockIdx.x * BN;

    float aSt[16];

    f32x8 acc[4][2];
#pragma unroll
    for (int s = 0; s < 4; ++s)
#pragma unroll
        for (int u = 0; u < 2; ++u)
            acc[s][u] = (f32x8){0.f, 0.f, 0.f, 0.f, 0.f, 0.f, 0.f, 0.f};

    // prologue: A regs for kt=0, async B copies for kt=0 into buffer 0
    load_a_stage(x, alpha, bBase, 0, t, aSt);
    stage_b_async(bhiG, bloG, sBhi[0], sBlo[0], nBase, 0, t);

    const int rb = t >> 3, ka = (t & 7) << 2;   // A store coords

    for (int kt = 0; kt < NKT; ++kt) {
        const int buf = kt & 1;

        // drain this wave's async B copies for buffer `buf`
        asm volatile("s_wait_asynccnt 0x0" ::: "memory");
        __syncthreads();  // previous compute done + everyone's copies done

        // ---- A registers -> LDS with fp32->bf16 hi/lo split ----
#pragma unroll
        for (int p = 0; p < 4; ++p) {
            U16x4 h, l;
            splitbf(aSt[p * 4 + 0], h.a, l.a);
            splitbf(aSt[p * 4 + 1], h.b, l.b);
            splitbf(aSt[p * 4 + 2], h.c, l.c);
            splitbf(aSt[p * 4 + 3], h.d, l.d);
            const int r = rb + p * 32;
            *(U16x4*)&sAhi[r * PITCH + ka] = h;
            *(U16x4*)&sAlo[r * PITCH + ka] = l;
        }
        __syncthreads();

        // ---- software pipeline: next tile (A sync loads, B async copies) ----
        if (kt + 1 < NKT) {
            load_a_stage(x, alpha, bBase, kt + 1, t, aSt);
            stage_b_async(bhiG, bloG, sBhi[buf ^ 1], sBlo[buf ^ 1], nBase, kt + 1, t);
        }

        // ---- build fragments per ISA 16-bit A(16x32)/B(32x16) layouts ----
        FragU ahi[4], alo[4], bhi[2], blo[2];
#pragma unroll
        for (int s = 0; s < 4; ++s) {
            const int m = waveMOff + s * 16 + ll;
            const u16* p1 = &sAhi[m * PITCH + lh * 8];
            ahi[s].q[0] = *(const uint4*)p1;
            ahi[s].q[1] = *(const uint4*)(p1 + 16);
            const u16* p2 = &sAlo[m * PITCH + lh * 8];
            alo[s].q[0] = *(const uint4*)p2;
            alo[s].q[1] = *(const uint4*)(p2 + 16);
        }
#pragma unroll
        for (int u = 0; u < 2; ++u) {
            const int n = waveNOff + u * 16 + ll;
            const u16* p1 = &sBhi[buf][n * PITCH + lh * 8];
            bhi[u].q[0] = *(const uint4*)p1;
            bhi[u].q[1] = *(const uint4*)(p1 + 16);
            const u16* p2 = &sBlo[buf][n * PITCH + lh * 8];
            blo[u].q[0] = *(const uint4*)p2;
            blo[u].q[1] = *(const uint4*)(p2 + 16);
        }

        // ---- bf16x3 WMMA: hi*hi + hi*lo + lo*hi ----
#pragma unroll
        for (int s = 0; s < 4; ++s) {
#pragma unroll
            for (int u = 0; u < 2; ++u) {
                acc[s][u] = __builtin_amdgcn_wmma_f32_16x16x32_bf16(
                    false, ahi[s].v, false, bhi[u].v, (short)0, acc[s][u], false, false);
                acc[s][u] = __builtin_amdgcn_wmma_f32_16x16x32_bf16(
                    false, ahi[s].v, false, blo[u].v, (short)0, acc[s][u], false, false);
                acc[s][u] = __builtin_amdgcn_wmma_f32_16x16x32_bf16(
                    false, alo[s].v, false, bhi[u].v, (short)0, acc[s][u], false, false);
            }
        }
    }

    // ---- epilogue: C/D 16x16 layout -> out[b,o] (bias folded into GEMM) ----
#pragma unroll
    for (int s = 0; s < 4; ++s) {
#pragma unroll
        for (int u = 0; u < 2; ++u) {
            const int o = nBase + waveNOff + u * 16 + ll;
#pragma unroll
            for (int j = 0; j < 8; ++j) {
                const int b = bBase + waveMOff + s * 16 + lh * 8 + j;
                out[(size_t)b * ODIM + o] = acc[s][u][j];
            }
        }
    }
}

// ---------------------------------------------------------------------------
extern "C" void kernel_launch(void* const* d_in, const int* in_sizes, int n_in,
                              void* d_out, int out_size, void* d_ws, size_t ws_size,
                              hipStream_t stream) {
    const float* x     = (const float*)d_in[0];  // [8192,1024]
    const float* phase = (const float*)d_in[1];  // [8192]
    const float* W     = (const float*)d_in[2];  // [6,1024,1024]
    const float* bias  = (const float*)d_in[3];  // [6,1024]
    float* out = (float*)d_out;                  // [8192,1024]

    // workspace layout: alpha (256 KB) | Bhi bf16 [1024][6176] | Blo bf16 [1024][6176]
    float* alpha = (float*)d_ws;
    u16*   bhi   = (u16*)((char*)d_ws + 256 * 1024);
    u16*   blo   = bhi + (size_t)ODIM * KPAD;

    pfnn_alpha_kernel<<<BATCH / 256, 256, 0, stream>>>(phase, alpha);
    pfnn_presplit_kernel<<<dim3((KPAD + 255) / 256, ODIM), 256, 0, stream>>>(W, bias, bhi, blo);
    pfnn_gemm_kernel<<<dim3(ODIM / BN, BATCH / BM), 256, 0, stream>>>(x, alpha, bhi, blo, out);
}
